// TeamMovementModel_56186762166636
// MI455X (gfx1250) — matrix operand order repaired and checked by reading, compiled, not run on hardware
//
#include <hip/hip_runtime.h>
#include <hip/hip_bf16.h>

typedef __attribute__((ext_vector_type(16))) _Float16 v16h;
typedef __attribute__((ext_vector_type(8)))  _Float16 v8h;
typedef __attribute__((ext_vector_type(8)))  float    v8f;
typedef __attribute__((ext_vector_type(2)))  float    f32x2;

constexpr int HID = 128;   // hidden size
constexpr int TT  = 128;   // timesteps
constexpr int SEQ_PER_BLK = 16;
constexpr int NPLAYER_SEQ = 64 * 22;   // 1408
constexpr int NBALL_SEQ   = 64;
constexpr int PLAYER_BLKS = NPLAYER_SEQ / SEQ_PER_BLK;  // 88
constexpr int BALL_BLKS   = NBALL_SEQ / SEQ_PER_BLK;    // 4

#if defined(__has_builtin) && __has_builtin(__builtin_amdgcn_tanhf)
__device__ __forceinline__ float tanh_hw(float x) {
    return __builtin_amdgcn_tanhf(x);            // v_tanh_f32 (gfx1250 TRANS op)
}
#else
__device__ __forceinline__ float tanh_hw(float x) {
    return 2.0f / (1.0f + __expf(-2.0f * x)) - 1.0f;
}
#endif

__device__ __forceinline__ float sigmoid_hw(float x) {
    // sigmoid(x) = 0.5*tanh(0.5x) + 0.5  -> mul + tanh + fma
    return fmaf(0.5f, tanh_hw(0.5f * x), 0.5f);
}

// One block = 16 sequences, 8 waves. Wave w owns hidden cols [16w,16w+16).
// Recurrent GEMM on v_wmma_f32_16x16x32_f16; W_hh lives in registers as
// preformatted B fragments for the whole T loop; h state double-buffers
// through LDS with a single barrier per timestep.
__launch_bounds__(256, 1)
__global__ void lstm_wmma_kernel(
    const float* __restrict__ x_players,  // [1408][128][2] (contiguous reshape)
    const float* __restrict__ x_ball,     // [64][128][2]
    const float* __restrict__ p_w_ih, const float* __restrict__ p_w_hh,
    const float* __restrict__ p_b_ih, const float* __restrict__ p_b_hh,
    const float* __restrict__ b_w_ih, const float* __restrict__ b_w_hh,
    const float* __restrict__ b_b_ih, const float* __restrict__ b_b_hh,
    float* __restrict__ hs_out)           // [1472][128] f32 (players then ball)
{
    __shared__ __align__(16) _Float16 s_h[2][SEQ_PER_BLK * HID];  // 2 x 4 KB
    __shared__ __align__(16) float    s_x[SEQ_PER_BLK * TT * 2];  // 16 KB

    const int tid  = threadIdx.x;
    const int lane = tid & 31;
    const int wave = tid >> 5;

    const bool isPlayer = (blockIdx.x < PLAYER_BLKS);
    const int  blkLocal = isPlayer ? blockIdx.x : (blockIdx.x - PLAYER_BLKS);
    const int  seq0     = isPlayer ? blkLocal * SEQ_PER_BLK
                                   : NPLAYER_SEQ + blkLocal * SEQ_PER_BLK;
    const float* x_base = (isPlayer ? x_players : x_ball)
                          + (size_t)blkLocal * SEQ_PER_BLK * TT * 2;
    const float* w_ih = isPlayer ? p_w_ih : b_w_ih;
    const float* w_hh = isPlayer ? p_w_hh : b_w_hh;
    const float* bih  = isPlayer ? p_b_ih : b_b_ih;
    const float* bhh  = isPlayer ? p_b_hh : b_b_hh;

    // Stage x tile (16 seq x 128 t x 2) into LDS; zero h state (buffer 0).
    for (int i = tid; i < SEQ_PER_BLK * TT * 2; i += 256) s_x[i] = x_base[i];
    for (int i = tid; i < SEQ_PER_BLK * HID;    i += 256) s_h[0][i] = (_Float16)0.0f;

    const int ncol = lane & 15;           // N within 16x16 tile
    const int nh   = wave * 16 + ncol;    // hidden column this lane produces
    const int selB = (lane < 16) ? 0 : 16;   // B K-half split
    const int selA = (lane < 16) ? 0 : 8;    // A K-quarter split
    const int moff = (lane < 16) ? 0 : 8;    // C/D row offset
    const int mA   = lane & 15;              // A row

    // Preformat B fragments of W_hh (f32 -> f16) once; resident across T loop.
    // B(32x16) layout: lane = N; elements = K[base..base+7], K[base+16..base+23], base = kc*32 + selA.
    v16h Bf[4][4];
    #pragma unroll
    for (int g = 0; g < 4; ++g) {
        const int nglob = g * HID + nh;
        #pragma unroll
        for (int kc = 0; kc < 4; ++kc) {
            const float* src = w_hh + (size_t)nglob * HID + kc * 32 + selA;
            v16h b;
            #pragma unroll
            for (int e = 0; e < 16; ++e) b[e] = (_Float16)src[e < 8 ? e : e + 8];
            Bf[g][kc] = b;
        }
    }

    // Per-lane input-projection constants (D=2) + merged bias.
    float wih0[4], wih1[4], bsum[4];
    #pragma unroll
    for (int g = 0; g < 4; ++g) {
        const int nglob = g * HID + nh;
        wih0[g] = w_ih[nglob * 2 + 0];
        wih1[g] = w_ih[nglob * 2 + 1];
        bsum[g] = bih[nglob] + bhh[nglob];
    }

    v8f c_state = {};   // cell state in C/D tile layout
    __syncthreads();

    for (int t = 0; t < TT; ++t) {
        const int rb = t & 1;        // h buffer to read
        const int wb = rb ^ 1;       // h buffer to write

        // x values for this lane's 8 C/D rows
        float x0[8], x1[8];
        #pragma unroll
        for (int v = 0; v < 8; ++v) {
            f32x2 xv = *(const f32x2*)&s_x[((moff + v) * TT + t) * 2];
            x0[v] = xv.x; x1[v] = xv.y;
        }
        // acc init = x @ W_ih^T + b_ih + b_hh  (folded input projection)
        v8f acc[4];
        #pragma unroll
        for (int g = 0; g < 4; ++g) {
            #pragma unroll
            for (int v = 0; v < 8; ++v)
                acc[g][v] = fmaf(x0[v], wih0[g], fmaf(x1[v], wih1[g], bsum[g]));
        }
        // acc += h @ W_hh^T   (4 K-chunks x 4 gates of 16x16x32 WMMA)
        #pragma unroll
        for (int kc = 0; kc < 4; ++kc) {
            const _Float16* hrow = &s_h[rb][mA * HID + kc * 32 + selA];
            v8h lo = *(const v8h*)(hrow);        // K[base .. base+7]
            v8h hi = *(const v8h*)(hrow + 16);   // K[base+16 .. base+23]
            v16h a = __builtin_shufflevector(lo, hi,
                       0,1,2,3,4,5,6,7,8,9,10,11,12,13,14,15);
            #pragma unroll
            for (int g = 0; g < 4; ++g) {
                acc[g] = __builtin_amdgcn_wmma_f32_16x16x32_f16(
                    false, a, false, Bf[g][kc], (short)0, acc[g], false, false);
            }
        }

        // gate nonlinearities + state update (TRANS/VALU, co-executes with XDL)
        v8f hnew;
        #pragma unroll
        for (int v = 0; v < 8; ++v) {
            float ig = sigmoid_hw(acc[0][v]);
            float fg = sigmoid_hw(acc[1][v]);
            float gg = tanh_hw(acc[2][v]);
            float og = sigmoid_hw(acc[3][v]);
            float cc = fmaf(fg, c_state[v], ig * gg);
            c_state[v] = cc;
            hnew[v] = og * tanh_hw(cc);
        }
        // publish h (f16) into the other buffer for next step's A fragments
        #pragma unroll
        for (int v = 0; v < 8; ++v)
            s_h[wb][(moff + v) * HID + nh] = (_Float16)hnew[v];

        if (t == TT - 1) {
            #pragma unroll
            for (int v = 0; v < 8; ++v)
                hs_out[(size_t)(seq0 + moff + v) * HID + nh] = hnew[v];
        }
        __syncthreads();   // one barrier/step: fences this step's reads + writes
    }
}

// out[b][p][j] = fc_b[j] + ph[b,p,:]·fc_w[j,0:128] + ball[b,:]·fc_w[j,128:256]
__global__ void fc_kernel(const float* __restrict__ hs,    // [1472][128]
                          const float* __restrict__ fc_w,  // [40][256]
                          const float* __restrict__ fc_b,  // [40]
                          float* __restrict__ out)         // [64*22*40]
{
    int idx = blockIdx.x * blockDim.x + threadIdx.x;
    if (idx >= 64 * 22 * 40) return;
    const int j  = idx % 40;
    const int bp = idx / 40;
    const int b  = bp / 22;
    const float* ph = hs + (size_t)bp * HID;
    const float* bh = hs + (size_t)(NPLAYER_SEQ + b) * HID;
    const float* w  = fc_w + (size_t)j * 256;
    float s = fc_b[j];
    #pragma unroll 4
    for (int k = 0; k < HID; ++k) s = fmaf(ph[k], w[k], s);
    #pragma unroll 4
    for (int k = 0; k < HID; ++k) s = fmaf(bh[k], w[HID + k], s);
    out[idx] = s;
}

extern "C" void kernel_launch(void* const* d_in, const int* in_sizes, int n_in,
                              void* d_out, int out_size, void* d_ws, size_t ws_size,
                              hipStream_t stream) {
    const float* x_players = (const float*)d_in[0];
    const float* x_ball    = (const float*)d_in[1];
    const float* p_w_ih    = (const float*)d_in[2];
    const float* p_w_hh    = (const float*)d_in[3];
    const float* p_b_ih    = (const float*)d_in[4];
    const float* p_b_hh    = (const float*)d_in[5];
    const float* b_w_ih    = (const float*)d_in[6];
    const float* b_w_hh    = (const float*)d_in[7];
    const float* b_b_ih    = (const float*)d_in[8];
    const float* b_b_hh    = (const float*)d_in[9];
    const float* fc_w      = (const float*)d_in[10];
    const float* fc_b      = (const float*)d_in[11];

    float* hs = (float*)d_ws;   // (1408 + 64) * 128 f32 final hidden states

    lstm_wmma_kernel<<<PLAYER_BLKS + BALL_BLKS, 256, 0, stream>>>(
        x_players, x_ball,
        p_w_ih, p_w_hh, p_b_ih, p_b_hh,
        b_w_ih, b_w_hh, b_b_ih, b_b_hh,
        hs);

    const int nout = 64 * 22 * 40;
    fc_kernel<<<(nout + 255) / 256, 256, 0, stream>>>(hs, fc_w, fc_b, (float*)d_out);
}
